// EdgeEncoder_81080392614045
// MI455X (gfx1250) — matrix operand order repaired and checked by reading, not compiled, not run on hardware
//
#include <hip/hip_runtime.h>
#include <hip/hip_bf16.h>

#define ZNODES 8192
#define KNBR   16
#define DMODEL 256
#define NPW    2           // nodes per workgroup
#define ROWS   (NPW*KNBR)  // 32 edge rows per workgroup (2 WMMA M-tiles)
#define SF     784         // feats LDS row stride (bf16 elems), 16B-aligned rows
#define SH     272         // h / Arbf LDS row stride (bf16 elems)
#define SPREAD2INV 0.64f   // 1/((20/16)^2)

typedef __attribute__((ext_vector_type(16))) __bf16 v16bf;
typedef __attribute__((ext_vector_type(8)))  __bf16 v8bf;
typedef __attribute__((ext_vector_type(8)))  float  v8f;

// ---------------------------------------------------------------------------
// Weight swizzle: row-major f32 W[kd x 256] -> fragment-major bf16.
// Tile (nt,kt) of 32(K)x16(N); within tile, lane L holds column n=L%16,
// K-range = kt*32 + (L/16)*16 + e  (e=0..15 contiguous) -> one v16bf per lane.
// ---------------------------------------------------------------------------
__global__ __launch_bounds__(256) void prep_weights_kernel(const float* __restrict__ W,
                                                           __bf16* __restrict__ out,
                                                           int kd) {
    int idx = blockIdx.x * 256 + threadIdx.x;
    int total = kd * DMODEL;
    if (idx >= total) return;
    int ktiles = kd >> 5;
    int e     = idx & 15;
    int lane  = (idx >> 4) & 31;
    int tflat = idx >> 9;                 // = nt*ktiles + kt
    int kt = tflat % ktiles;
    int nt = tflat / ktiles;
    int k = kt * 32 + (lane >> 4) * 16 + e;
    int n = nt * 16 + (lane & 15);
    out[idx] = (__bf16)W[k * DMODEL + n];
}

// Pack (Ca.xyz, sample) into one float4 per node for the coalesced KNN scan.
__global__ __launch_bounds__(256) void prep_ca_kernel(const float* __restrict__ coords,
                                                      const int* __restrict__ sample,
                                                      float4* __restrict__ capk) {
    int i = blockIdx.x * 256 + threadIdx.x;
    if (i >= ZNODES) return;
    float4 v;
    v.x = coords[(i * 4 + 1) * 3 + 0];
    v.y = coords[(i * 4 + 1) * 3 + 1];
    v.z = coords[(i * 4 + 1) * 3 + 2];
    v.w = __int_as_float(sample[i]);
    capk[i] = v;
}

// ---------------------------------------------------------------------------
// KNN: one wave per node. Per-lane sorted top-16 over a strided 1/32 slice
// (one b128 load per candidate), then a 32-way shuffle-merge (16 rounds).
// ---------------------------------------------------------------------------
__global__ __launch_bounds__(256) void knn_kernel(const float4* __restrict__ capk,
                                                  int* __restrict__ nbrs_i,
                                                  float* __restrict__ out_nbrs,
                                                  float* __restrict__ out_mask) {
    __shared__ float sd[8 * 512];
    __shared__ int   si[8 * 512];
    int tid = threadIdx.x, lane = tid & 31, wave = tid >> 5;
    int i = blockIdx.x * 8 + wave;

    float4 ci = capk[i];
    int smp = __float_as_int(ci.w);

    float bd[16]; int bi[16];
#pragma unroll
    for (int t = 0; t < 16; ++t) { bd[t] = 3.4e38f; bi[t] = 0x7fffffff; }

    for (int j = lane; j < ZNODES; j += 32) {
        float4 cj = capk[j];
        float dx = cj.x - ci.x, dy = cj.y - ci.y, dz = cj.z - ci.z;
        float d2 = dx * dx + dy * dy + dz * dz;
        if (__float_as_int(cj.w) != smp) d2 = 3.0e38f;
        if (d2 < bd[15] || (d2 == bd[15] && j < bi[15])) {
            bd[15] = d2; bi[15] = j;
#pragma unroll
            for (int s = 15; s > 0; --s) {
                bool sw = (bd[s] < bd[s - 1]) || (bd[s] == bd[s - 1] && bi[s] < bi[s - 1]);
                float td = bd[s]; int ti = bi[s];
                if (sw) { bd[s] = bd[s - 1]; bi[s] = bi[s - 1]; bd[s - 1] = td; bi[s - 1] = ti; }
            }
        }
    }
#pragma unroll
    for (int t = 0; t < 16; ++t) {
        sd[wave * 512 + lane * 16 + t] = bd[t];
        si[wave * 512 + lane * 16 + t] = bi[t];
    }
    __syncthreads();

    int ptr = 0;
    for (int sel = 0; sel < 16; ++sel) {
        float d  = (ptr < 16) ? sd[wave * 512 + lane * 16 + ptr] : 3.4e38f;
        int  idv = (ptr < 16) ? si[wave * 512 + lane * 16 + ptr] : 0x7fffffff;
        int  ln  = lane;
        for (int off = 16; off; off >>= 1) {
            float od = __shfl_xor(d, off, 32);
            int   oi = __shfl_xor(idv, off, 32);
            int   ol = __shfl_xor(ln, off, 32);
            if (od < d || (od == d && oi < idv)) { d = od; idv = oi; ln = ol; }
        }
        if (lane == ln) ptr++;
        if (lane == 0) {
            bool ok = d < 1.0e37f;
            int nb = ok ? idv : i;
            nbrs_i[i * 16 + sel]   = nb;
            out_nbrs[i * 16 + sel] = (float)nb;
            out_mask[i * 16 + sel] = ok ? 1.0f : 0.0f;
        }
    }
}

// ---------------------------------------------------------------------------
// 2x2 register-blocked WMMA macro-tile: each wave computes output tiles
// (mt,nt) in {0,1} x {nt0,nt1}. Each A/B fragment feeds two WMMAs ->
// halves LDS + L2 traffic per FLOP versus a 1x1 tile loop.
// ---------------------------------------------------------------------------
__device__ __forceinline__ v16bf ldsA_frag(const __bf16* p) {
    v8bf c0 = *(const v8bf*)(p);
    v8bf c1 = *(const v8bf*)(p + 8);
    v16bf a;
#pragma unroll
    for (int e = 0; e < 8; ++e) { a[e] = c0[e]; a[e + 8] = c1[e]; }
    return a;
}

__device__ __forceinline__ void wmma_2x2(const __bf16* A, int lda,
                                         const __bf16* __restrict__ Wsw,
                                         int ktiles, int nt0, int nt1, int lane,
                                         v8f& c00, v8f& c01, v8f& c10, v8f& c11) {
    int lane16 = lane & 15, khalf = lane >> 4;
    const __bf16* ar0 = A + lane16 * lda + khalf * 16;
    const __bf16* ar1 = A + (16 + lane16) * lda + khalf * 16;
    const __bf16* w0 = Wsw + (size_t)nt0 * ktiles * 512 + (size_t)lane * 16;
    const __bf16* w1 = Wsw + (size_t)nt1 * ktiles * 512 + (size_t)lane * 16;
    for (int kt = 0; kt < ktiles; ++kt) {
        v16bf b0 = *(const v16bf*)(w0 + (size_t)kt * 512);
        v16bf b1 = *(const v16bf*)(w1 + (size_t)kt * 512);
        v16bf a0 = ldsA_frag(ar0 + kt * 32);
        v16bf a1 = ldsA_frag(ar1 + kt * 32);
        c00 = __builtin_amdgcn_wmma_f32_16x16x32_bf16(false, a0, false, b0, (short)0, c00, false, false);
        c01 = __builtin_amdgcn_wmma_f32_16x16x32_bf16(false, a0, false, b1, (short)0, c01, false, false);
        c10 = __builtin_amdgcn_wmma_f32_16x16x32_bf16(false, a1, false, b0, (short)0, c10, false, false);
        c11 = __builtin_amdgcn_wmma_f32_16x16x32_bf16(false, a1, false, b1, (short)0, c11, false, false);
    }
}

#define ZERO8F {0.f,0.f,0.f,0.f,0.f,0.f,0.f,0.f}

// ---------------------------------------------------------------------------
// Edge encoder: 2 nodes (32 edge rows) per 256-thread workgroup.
// ---------------------------------------------------------------------------
__global__ __launch_bounds__(256) void edge_kernel(
    const float* __restrict__ coords, const float* __restrict__ frames,
    const int* __restrict__ seq_pos, const int* __restrict__ chain_pos,
    const float* __restrict__ centers,
    const float* __restrict__ ln_rbf_g, const float* __restrict__ ln_rbf_b,
    const float* __restrict__ b_rbf, const float* __restrict__ Wfr,
    const float* __restrict__ b_frame, const float* __restrict__ seq_emb,
    const float* __restrict__ ln_e_g, const float* __restrict__ ln_e_b,
    const float* __restrict__ b1, const float* __restrict__ b2,
    const float* __restrict__ b3,
    const __bf16* __restrict__ Wrbf_sw, const __bf16* __restrict__ W1_sw,
    const __bf16* __restrict__ W2_sw,  const __bf16* __restrict__ W3_sw,
    const int* __restrict__ nbrs, float* __restrict__ out) {

    __shared__ __bf16 feats[ROWS * SF];   // 768-wide concat features (bf16)
    __shared__ __bf16 Arbf[ROWS * SH];    // LN(rbf) GEMM input
    __shared__ __bf16 h1[ROWS * SH];
    __shared__ __bf16 h2[ROWS * SH];
    __shared__ float  wfr_s[9 * DMODEL];
    __shared__ float  relf[ROWS * 12];
    __shared__ float  distb[ROWS * 16];
    __shared__ int    nbb[ROWS];
    __shared__ int    relidx[ROWS];

    int tid = threadIdx.x, lane = tid & 31, wave = tid >> 5;
    int lane16 = lane & 15, khalf = lane >> 4;
    int node0 = blockIdx.x * NPW;
    int nt0 = wave, nt1 = wave + 8;

    // phase 0: W_frame -> LDS, per-edge scalars (nbr, rel-seq-idx, rel-frame 3x3)
    for (int t = tid; t < 9 * DMODEL; t += 256) wfr_s[t] = Wfr[t];
    if (tid < ROWS) {
        int r = tid, node = node0 + (r >> 4);
        int nb = nbrs[node * KNBR + (r & 15)];
        nbb[r] = nb;
        int rel = seq_pos[nb] - seq_pos[node];
        rel = min(32, max(-32, rel));
        if (chain_pos[nb] != chain_pos[node]) rel = 33;
        relidx[r] = rel + 32;
        const float* Fi = frames + node * 9;
        const float* Fn = frames + nb * 9;
#pragma unroll
        for (int b = 0; b < 3; ++b)
#pragma unroll
            for (int c = 0; c < 3; ++c) {
                float s = 0.f;
#pragma unroll
                for (int a = 0; a < 3; ++a) s += Fi[a * 3 + b] * Fn[a * 3 + c];
                relf[r * 12 + b * 3 + c] = s;
            }
    }
    __syncthreads();

    // phase 1: 4x4 atom-pair distances per edge
    for (int t = tid; t < ROWS * 16; t += 256) {
        int r = t >> 4, ab = t & 15, a = ab >> 2, b = ab & 3;
        int node = node0 + (r >> 4), nb = nbb[r];
        float s = 0.f;
#pragma unroll
        for (int d = 0; d < 3; ++d) {
            float df = coords[(node * 4 + a) * 3 + d] - coords[(nb * 4 + b) * 3 + d];
            s += df * df;
        }
        distb[t] = sqrtf(s);
    }
    __syncthreads();

    // phase 2: raw RBF -> feats[:, 0:256] (temp storage)
    for (int idx = tid; idx < ROWS * 256; idx += 256) {
        int r = idx >> 8, f = idx & 255;
        float t = distb[r * 16 + (f >> 4)] - centers[f & 15];
        feats[r * SF + f] = (__bf16)__expf(-t * t * SPREAD2INV);
    }
    __syncthreads();

    // phase 3: LN(256) over raw RBF -> Arbf (wave handles 4 rows)
    for (int rr = 0; rr < 4; ++rr) {
        int r = wave * 4 + rr;
        float s = 0.f, s2 = 0.f;
        for (int c = lane; c < 256; c += 32) {
            float x = (float)feats[r * SF + c]; s += x; s2 += x * x;
        }
        for (int off = 16; off; off >>= 1) { s += __shfl_xor(s, off, 32); s2 += __shfl_xor(s2, off, 32); }
        float m = s * (1.f / 256.f), v = s2 * (1.f / 256.f) - m * m;
        float rs = rsqrtf(v + 1e-5f);
        for (int c = lane; c < 256; c += 32) {
            float x = (float)feats[r * SF + c];
            Arbf[r * SH + c] = (__bf16)((x - m) * rs * ln_rbf_g[c] + ln_rbf_b[c]);
        }
    }
    __syncthreads();

    // phase 4: rel_rbf = LN(rbf) @ W_rbf + b_rbf -> feats[:, 0:256]
    {
        v8f c00 = ZERO8F, c01 = ZERO8F, c10 = ZERO8F, c11 = ZERO8F;
        wmma_2x2(Arbf, SH, Wrbf_sw, 8, nt0, nt1, lane, c00, c01, c10, c11);
        int col0 = nt0 * 16 + lane16, col1 = nt1 * 16 + lane16;
        float bv0 = b_rbf[col0], bv1 = b_rbf[col1];
#pragma unroll
        for (int r8 = 0; r8 < 8; ++r8) {
            int row0 = khalf * 8 + r8, row1 = 16 + khalf * 8 + r8;
            feats[row0 * SF + col0] = (__bf16)(c00[r8] + bv0);
            feats[row0 * SF + col1] = (__bf16)(c01[r8] + bv1);
            feats[row1 * SF + col0] = (__bf16)(c10[r8] + bv0);
            feats[row1 * SF + col1] = (__bf16)(c11[r8] + bv1);
        }
    }
    // phase 5: frames 9->256 matvec -> feats[:,256:512]; embedding -> feats[:,512:768]
    for (int idx = tid; idx < ROWS * 256; idx += 256) {
        int r = idx >> 8, n = idx & 255;
        float s = b_frame[n];
#pragma unroll
        for (int q = 0; q < 9; ++q) s += relf[r * 12 + q] * wfr_s[q * DMODEL + n];
        feats[r * SF + 256 + n] = (__bf16)s;
        feats[r * SF + 512 + n] = (__bf16)seq_emb[relidx[r] * DMODEL + n];
    }
    __syncthreads();

    // phase 6: LN(768) in place (wave owns its 4 rows)
    for (int rr = 0; rr < 4; ++rr) {
        int r = wave * 4 + rr;
        float s = 0.f, s2 = 0.f;
        for (int c = lane; c < 768; c += 32) {
            float x = (float)feats[r * SF + c]; s += x; s2 += x * x;
        }
        for (int off = 16; off; off >>= 1) { s += __shfl_xor(s, off, 32); s2 += __shfl_xor(s2, off, 32); }
        float m = s * (1.f / 768.f), v = s2 * (1.f / 768.f) - m * m;
        float rs = rsqrtf(v + 1e-5f);
        for (int c = lane; c < 768; c += 32) {
            float x = (float)feats[r * SF + c];
            feats[r * SF + c] = (__bf16)((x - m) * rs * ln_e_g[c] + ln_e_b[c]);
        }
    }
    __syncthreads();

    // phase 7: h1 = silu(feats @ W1 + b1), K = 768
    {
        v8f c00 = ZERO8F, c01 = ZERO8F, c10 = ZERO8F, c11 = ZERO8F;
        wmma_2x2(feats, SF, W1_sw, 24, nt0, nt1, lane, c00, c01, c10, c11);
        int col0 = nt0 * 16 + lane16, col1 = nt1 * 16 + lane16;
        float bv0 = b1[col0], bv1 = b1[col1];
#pragma unroll
        for (int r8 = 0; r8 < 8; ++r8) {
            int row0 = khalf * 8 + r8, row1 = 16 + khalf * 8 + r8;
            float x;
            x = c00[r8] + bv0; h1[row0 * SH + col0] = (__bf16)(x / (1.f + __expf(-x)));
            x = c01[r8] + bv1; h1[row0 * SH + col1] = (__bf16)(x / (1.f + __expf(-x)));
            x = c10[r8] + bv0; h1[row1 * SH + col0] = (__bf16)(x / (1.f + __expf(-x)));
            x = c11[r8] + bv1; h1[row1 * SH + col1] = (__bf16)(x / (1.f + __expf(-x)));
        }
    }
    __syncthreads();

    // phase 8: h2 = silu(h1 @ W2 + b2), K = 256
    {
        v8f c00 = ZERO8F, c01 = ZERO8F, c10 = ZERO8F, c11 = ZERO8F;
        wmma_2x2(h1, SH, W2_sw, 8, nt0, nt1, lane, c00, c01, c10, c11);
        int col0 = nt0 * 16 + lane16, col1 = nt1 * 16 + lane16;
        float bv0 = b2[col0], bv1 = b2[col1];
#pragma unroll
        for (int r8 = 0; r8 < 8; ++r8) {
            int row0 = khalf * 8 + r8, row1 = 16 + khalf * 8 + r8;
            float x;
            x = c00[r8] + bv0; h2[row0 * SH + col0] = (__bf16)(x / (1.f + __expf(-x)));
            x = c01[r8] + bv1; h2[row0 * SH + col1] = (__bf16)(x / (1.f + __expf(-x)));
            x = c10[r8] + bv0; h2[row1 * SH + col0] = (__bf16)(x / (1.f + __expf(-x)));
            x = c11[r8] + bv1; h2[row1 * SH + col1] = (__bf16)(x / (1.f + __expf(-x)));
        }
    }
    __syncthreads();

    // phase 9: edges = h2 @ W3 + b3 -> global f32
    {
        v8f c00 = ZERO8F, c01 = ZERO8F, c10 = ZERO8F, c11 = ZERO8F;
        wmma_2x2(h2, SH, W3_sw, 8, nt0, nt1, lane, c00, c01, c10, c11);
        int col0 = nt0 * 16 + lane16, col1 = nt1 * 16 + lane16;
        float bv0 = b3[col0], bv1 = b3[col1];
        float* ob = out + (size_t)blockIdx.x * ROWS * DMODEL;
#pragma unroll
        for (int r8 = 0; r8 < 8; ++r8) {
            int row0 = khalf * 8 + r8, row1 = 16 + khalf * 8 + r8;
            ob[row0 * DMODEL + col0] = c00[r8] + bv0;
            ob[row0 * DMODEL + col1] = c01[r8] + bv1;
            ob[row1 * DMODEL + col0] = c10[r8] + bv0;
            ob[row1 * DMODEL + col1] = c11[r8] + bv1;
        }
    }
}

extern "C" void kernel_launch(void* const* d_in, const int* in_sizes, int n_in,
                              void* d_out, int out_size, void* d_ws, size_t ws_size,
                              hipStream_t stream) {
    (void)in_sizes; (void)n_in; (void)out_size; (void)ws_size;
    const float* coords    = (const float*)d_in[0];
    const float* frames    = (const float*)d_in[1];
    const int*   seq_pos   = (const int*)d_in[2];
    const int*   chain_pos = (const int*)d_in[3];
    const int*   sample    = (const int*)d_in[4];
    const float* centers   = (const float*)d_in[5];
    const float* ln_rbf_g  = (const float*)d_in[6];
    const float* ln_rbf_b  = (const float*)d_in[7];
    const float* W_rbf     = (const float*)d_in[8];
    const float* b_rbf     = (const float*)d_in[9];
    const float* W_frame   = (const float*)d_in[10];
    const float* b_frame   = (const float*)d_in[11];
    const float* seq_emb   = (const float*)d_in[12];
    const float* ln_e_g    = (const float*)d_in[13];
    const float* ln_e_b    = (const float*)d_in[14];
    const float* W1        = (const float*)d_in[15];
    const float* b1        = (const float*)d_in[16];
    const float* W2        = (const float*)d_in[17];
    const float* b2        = (const float*)d_in[18];
    const float* W3        = (const float*)d_in[19];
    const float* b3        = (const float*)d_in[20];

    char* ws = (char*)d_ws;
    int*    nbrs_i  = (int*)ws;                       // 8192*16*4 = 524288 B
    __bf16* Wrbf_sw = (__bf16*)(ws + 524288);         // 131072 B
    __bf16* W1_sw   = (__bf16*)(ws + 655360);         // 393216 B
    __bf16* W2_sw   = (__bf16*)(ws + 1048576);        // 131072 B
    __bf16* W3_sw   = (__bf16*)(ws + 1179648);        // 131072 B
    float4* capk    = (float4*)(ws + 1310720);        // 8192*16 = 131072 B

    float* out       = (float*)d_out;                 // edges [8192*16*256]
    float* out_nbrs  = out + (size_t)ZNODES * KNBR * DMODEL;
    float* out_mask  = out_nbrs + (size_t)ZNODES * KNBR;

    // weight swizzle (L2-resident bf16, fragment-major) + packed Ca array
    prep_weights_kernel<<<(256 * DMODEL + 255) / 256, 256, 0, stream>>>(W_rbf, Wrbf_sw, 256);
    prep_weights_kernel<<<(768 * DMODEL + 255) / 256, 256, 0, stream>>>(W1,    W1_sw,   768);
    prep_weights_kernel<<<(256 * DMODEL + 255) / 256, 256, 0, stream>>>(W2,    W2_sw,   256);
    prep_weights_kernel<<<(256 * DMODEL + 255) / 256, 256, 0, stream>>>(W3,    W3_sw,   256);
    prep_ca_kernel<<<ZNODES / 256, 256, 0, stream>>>(coords, sample, capk);

    // KNN: one wave per node, 8 waves per block
    knn_kernel<<<ZNODES / 8, 256, 0, stream>>>(capk, nbrs_i, out_nbrs, out_mask);

    // Edge encoder: 2 nodes per workgroup
    edge_kernel<<<ZNODES / NPW, 256, 0, stream>>>(
        coords, frames, seq_pos, chain_pos, centers, ln_rbf_g, ln_rbf_b,
        b_rbf, W_frame, b_frame, seq_emb, ln_e_g, ln_e_b, b1, b2, b3,
        Wrbf_sw, W1_sw, W2_sw, W3_sw, nbrs_i, out);
}
